// EncoderBlock_91207925498450
// MI455X (gfx1250) — matrix operand order repaired and checked by reading, not compiled, not run on hardware
//
#include <hip/hip_runtime.h>
#include <hip/hip_bf16.h>

// Transformer encoder block (B=1, S=4096, D=768, H=12, DFF=3072) on gfx1250.
// All GEMMs via v_wmma_f32_16x16x32_bf16 (fp32 accumulate), flash-style attention.
// Row-major LDS tile staging uses GLOBAL_LOAD_ASYNC_TO_LDS_B128 when available.

typedef __bf16 bf16;
typedef __attribute__((ext_vector_type(16))) __bf16 v16bf;
typedef __attribute__((ext_vector_type(8)))  float  v8f;

#define SEQ    4096
#define DMODEL 768
#define DFF    3072
#define NH     12
#define DHEAD  64
#define DQKV   2304   // Q|K|V packed

#if defined(__has_builtin)
#if __has_builtin(__builtin_amdgcn_global_load_async_to_lds_b128) && \
    __has_builtin(__builtin_amdgcn_s_wait_asynccnt)
#define USE_ASYNC_LDS 1
#endif
#endif

#if USE_ASYNC_LDS
typedef int v4i __attribute__((vector_size(16)));
typedef __attribute__((address_space(1))) v4i gv4i;   // global, printed as __device__
typedef __attribute__((address_space(3))) v4i sv4i;   // LDS, printed as __shared__
__device__ __forceinline__ void async_copy_b128(const void* g, void* l) {
  __builtin_amdgcn_global_load_async_to_lds_b128((gv4i*)g, (sv4i*)l, 0, 0);
}
__device__ __forceinline__ void async_wait0() {
  __builtin_amdgcn_s_wait_asynccnt(0);
}
#else
__device__ __forceinline__ void async_copy_b128(const void* g, void* l) {
  *(uint4*)l = *(const uint4*)g;
}
__device__ __forceinline__ void async_wait0() {}
#endif

__device__ __forceinline__ bf16 f2bf(float f) {
  union { float f; unsigned u; } x; x.f = f;
  unsigned r = (x.u + 0x7fffu + ((x.u >> 16) & 1u)) >> 16;  // RNE
  union { unsigned short s; bf16 b; } y; y.s = (unsigned short)r;
  return y.b;
}

struct FragBF { union { v16bf v; unsigned u[8]; }; };

__device__ __forceinline__ v8f wmma_bf16(const FragBF& a, const FragBF& b, v8f c) {
  return __builtin_amdgcn_wmma_f32_16x16x32_bf16(false, a.v, false, b.v,
                                                 (short)0, c, false, false);
}

// A-matrix 16x32 bf16 fragment from row-major [m][k] storage (contiguous K order, same as load_b_frag):
// lane<16: M=lane, K=0..15; lane>=16: same M, K=16..31 (2 K per VGPR).
__device__ __forceinline__ void load_a_frag(FragBF& fr, const bf16* base, int stride, int lane) {
  const int half = lane >> 4, m = lane & 15;
  const bf16* row = base + m * stride;
  #pragma unroll
  for (int v = 0; v < 8; ++v) {
    int kb = half * 16 + 2 * v;
    fr.u[v] = *(const unsigned*)(row + kb);
  }
}

// B-matrix 32x16 bf16 fragment from storage where element [n][k] is at
// base + n*stride + k (K contiguous per column). lane<16: N=lane, K=0..15;
// lane>=16: same N, K=16..31 (2 K per VGPR).
__device__ __forceinline__ void load_b_frag(FragBF& fr, const bf16* base, int stride, int lane) {
  const int half = lane >> 4, n = lane & 15;
  const bf16* row = base + n * stride;
  #pragma unroll
  for (int v = 0; v < 8; ++v) {
    int k = half * 16 + 2 * v;
    fr.u[v] = *(const unsigned*)(row + k);
  }
}

// ---------------- conversion / packing ----------------

__global__ void k_cvt(const float* __restrict__ s, bf16* __restrict__ d, int n) {
  int i = blockIdx.x * blockDim.x + threadIdx.x;
  if (i < n) d[i] = f2bf(s[i]);
}

__global__ void k_pack_qkv(const float* __restrict__ wq, const float* __restrict__ wk,
                           const float* __restrict__ wv,
                           const float* __restrict__ bq, const float* __restrict__ bk,
                           const float* __restrict__ bv,
                           bf16* __restrict__ w, float* __restrict__ b) {
  int i = blockIdx.x * blockDim.x + threadIdx.x;
  if (i < DMODEL * DMODEL) {
    int r = i / DMODEL, c = i % DMODEL;
    w[(size_t)r * DQKV + c]              = f2bf(wq[i]);
    w[(size_t)r * DQKV + DMODEL + c]     = f2bf(wk[i]);
    w[(size_t)r * DQKV + 2 * DMODEL + c] = f2bf(wv[i]);
  }
  if (i < DMODEL) { b[i] = bq[i]; b[DMODEL + i] = bk[i]; b[2 * DMODEL + i] = bv[i]; }
}

// ---------------- WMMA GEMM: C = A(MxK) @ W(KxN) + bias [+res] [relu] ----------------
// Block tile 128x128, BK=32, 256 threads = 8 waves, each wave a 32x64 sub-tile
// (2 A-frags x 4 B-frags = 8 WMMAs per K-step).

template <bool RELU, bool RES, bool WF32, bool WBF16>
__global__ __launch_bounds__(256, 2)
void k_gemm(const bf16* __restrict__ A, const bf16* __restrict__ W,
            const float* __restrict__ bias, const float* __restrict__ res,
            float* __restrict__ outF, bf16* __restrict__ outB,
            int M, int N, int K) {
  constexpr int BM = 128, BN = 128, BK = 32, AS = BK + 8, BS = BK + 8;
  __shared__ bf16 As[BM][AS];    // [m][k] row-major
  __shared__ bf16 Bst[BN][BS];   // [n][k] transposed so B-frag K-pairs are contiguous

  const int tid = threadIdx.x;
  const int wave = tid >> 5, lane = tid & 31;
  const int wm = (wave & 3) * 32, wn = (wave >> 2) * 64;
  const int m0 = blockIdx.y * BM, n0 = blockIdx.x * BN;

  v8f acc[2][4];
  #pragma unroll
  for (int i = 0; i < 2; ++i)
    #pragma unroll
    for (int j = 0; j < 4; ++j)
      #pragma unroll
      for (int r = 0; r < 8; ++r) acc[i][j][r] = 0.f;

  for (int k0 = 0; k0 < K; k0 += BK) {
    __syncthreads();
    // A tile 128x32: 512 x b128, 2 per thread (async when available)
    #pragma unroll
    for (int p = 0; p < 2; ++p) {
      int idx = p * 256 + tid;
      int r = idx >> 2, c = (idx & 3) * 8;
      async_copy_b128(A + (size_t)(m0 + r) * K + k0 + c, &As[r][c]);
    }
    // W tile 32x128 -> transposed into Bst (VGPR scatter, 2 x b128 per thread)
    #pragma unroll
    for (int p = 0; p < 2; ++p) {
      int idx = p * 256 + tid;
      int r = idx >> 4, c = (idx & 15) * 8;
      union { uint4 q; bf16 h[8]; } v;
      v.q = *(const uint4*)(W + (size_t)(k0 + r) * N + n0 + c);
      #pragma unroll
      for (int j = 0; j < 8; ++j) Bst[c + j][r] = v.h[j];
    }
    async_wait0();
    __syncthreads();

    FragBF a[2], b[4];
    #pragma unroll
    for (int i = 0; i < 2; ++i) load_a_frag(a[i], &As[wm + i * 16][0], AS, lane);
    #pragma unroll
    for (int j = 0; j < 4; ++j) load_b_frag(b[j], &Bst[wn + j * 16][0], BS, lane);
    #pragma unroll
    for (int i = 0; i < 2; ++i)
      #pragma unroll
      for (int j = 0; j < 4; ++j)
        acc[i][j] = wmma_bf16(a[i], b[j], acc[i][j]);
  }

  const int half = lane >> 4, l16 = lane & 15;
  #pragma unroll
  for (int i = 0; i < 2; ++i) {
    #pragma unroll
    for (int j = 0; j < 4; ++j) {
      const int col = n0 + wn + j * 16 + l16;
      const float bi = bias[col];
      #pragma unroll
      for (int r = 0; r < 8; ++r) {
        const int row = m0 + wm + i * 16 + half * 8 + r;
        float v = acc[i][j][r] + bi;
        if (RES)  v += res[(size_t)row * N + col];
        if (RELU) v = v > 0.f ? v : 0.f;
        if (WF32)  outF[(size_t)row * N + col] = v;
        if (WBF16) outB[(size_t)row * N + col] = f2bf(v);
      }
    }
  }
}

// ---------------- flash attention ----------------
// Grid: (SEQ/128, NH). 8 waves; wave owns 16 query rows. Key tiles of 64.

__global__ __launch_bounds__(256, 2)
void k_attn(const bf16* __restrict__ qkv, const int* __restrict__ mask,
            bf16* __restrict__ ctx) {
  __shared__ bf16 Ks[64][72];      // [key][d]
  __shared__ bf16 Vst[64][72];     // [d][key] (transposed)
  __shared__ bf16 Ps[8][16][72];   // per-wave P staging [qrow][key]

  const int tid = threadIdx.x, wave = tid >> 5, lane = tid & 31;
  const int half = lane >> 4, l16 = lane & 15;
  const int head = blockIdx.y;
  const int qr0 = blockIdx.x * 128 + wave * 16;
  const int hq = head * DHEAD;

  FragBF aq[2];
  load_a_frag(aq[0], qkv + (size_t)qr0 * DQKV + hq,      DQKV, lane);
  load_a_frag(aq[1], qkv + (size_t)qr0 * DQKV + hq + 32, DQKV, lane);

  float mrow[8], lrow[8];
  v8f accO[4];
  #pragma unroll
  for (int r = 0; r < 8; ++r) { mrow[r] = -3.0e38f; lrow[r] = 0.f; }
  #pragma unroll
  for (int t = 0; t < 4; ++t)
    #pragma unroll
    for (int r = 0; r < 8; ++r) accO[t][r] = 0.f;

  for (int k0 = 0; k0 < SEQ; k0 += 64) {
    __syncthreads();
    // stage K (row-major, async-capable) and V (transposed): 64x64 each
    #pragma unroll
    for (int p = 0; p < 2; ++p) {
      int idx = p * 256 + tid;
      int r = idx >> 3, c = (idx & 7) * 8;
      async_copy_b128(qkv + (size_t)(k0 + r) * DQKV + DMODEL + hq + c, &Ks[r][c]);
      union { uint4 q; bf16 h[8]; } v;
      v.q = *(const uint4*)(qkv + (size_t)(k0 + r) * DQKV + 2 * DMODEL + hq + c);
      #pragma unroll
      for (int j = 0; j < 8; ++j) Vst[c + j][r] = v.h[j];
    }
    async_wait0();
    __syncthreads();

    // scores S = Q @ K^T / sqrt(dk), masked
    v8f sc[4];
    #pragma unroll
    for (int nt = 0; nt < 4; ++nt) {
      FragBF b0, b1;
      load_b_frag(b0, &Ks[nt * 16][0],  72, lane);
      load_b_frag(b1, &Ks[nt * 16][32], 72, lane);
      v8f s;
      #pragma unroll
      for (int r = 0; r < 8; ++r) s[r] = 0.f;
      s = wmma_bf16(aq[0], b0, s);
      s = wmma_bf16(aq[1], b1, s);
      const int key = k0 + nt * 16 + l16;
      #pragma unroll
      for (int r = 0; r < 8; ++r) {
        s[r] *= 0.125f;  // 1/sqrt(64)
        const int qrow = qr0 + half * 8 + r;
        if (mask[(size_t)qrow * SEQ + key] == 0) s[r] = -1.0e9f;
      }
      sc[nt] = s;
    }

    // online softmax (row stats live in the 16-lane half owning that row)
    float corr[8];
    #pragma unroll
    for (int r = 0; r < 8; ++r) {
      float mx = fmaxf(fmaxf(sc[0][r], sc[1][r]), fmaxf(sc[2][r], sc[3][r]));
      mx = fmaxf(mx, __shfl_xor(mx, 1));
      mx = fmaxf(mx, __shfl_xor(mx, 2));
      mx = fmaxf(mx, __shfl_xor(mx, 4));
      mx = fmaxf(mx, __shfl_xor(mx, 8));
      const float mnew = fmaxf(mrow[r], mx);
      corr[r] = __expf(mrow[r] - mnew);
      mrow[r] = mnew;
      float rs = 0.f;
      #pragma unroll
      for (int nt = 0; nt < 4; ++nt) {
        float p = __expf(sc[nt][r] - mnew);
        sc[nt][r] = p;
        rs += p;
      }
      rs += __shfl_xor(rs, 1);
      rs += __shfl_xor(rs, 2);
      rs += __shfl_xor(rs, 4);
      rs += __shfl_xor(rs, 8);
      lrow[r] = lrow[r] * corr[r] + rs;
      #pragma unroll
      for (int t = 0; t < 4; ++t) accO[t][r] *= corr[r];
    }

    // P: C-frag layout -> A-frag layout via per-wave LDS (DS ops in-order per wave)
    #pragma unroll
    for (int nt = 0; nt < 4; ++nt)
      #pragma unroll
      for (int r = 0; r < 8; ++r)
        Ps[wave][half * 8 + r][nt * 16 + l16] = f2bf(sc[nt][r]);

    FragBF ap0, ap1;
    load_a_frag(ap0, &Ps[wave][0][0],  72, lane);
    load_a_frag(ap1, &Ps[wave][0][32], 72, lane);
    #pragma unroll
    for (int dt = 0; dt < 4; ++dt) {
      FragBF bv0, bv1;
      load_b_frag(bv0, &Vst[dt * 16][0],  72, lane);
      load_b_frag(bv1, &Vst[dt * 16][32], 72, lane);
      accO[dt] = wmma_bf16(ap0, bv0, accO[dt]);
      accO[dt] = wmma_bf16(ap1, bv1, accO[dt]);
    }
  }

  #pragma unroll
  for (int r = 0; r < 8; ++r) {
    const float inv = 1.0f / lrow[r];
    const int qrow = qr0 + half * 8 + r;
    #pragma unroll
    for (int dt = 0; dt < 4; ++dt)
      ctx[(size_t)qrow * DMODEL + hq + dt * 16 + l16] = f2bf(accO[dt][r] * inv);
  }
}

// ---------------- launcher ----------------

extern "C" void kernel_launch(void* const* d_in, const int* in_sizes, int n_in,
                              void* d_out, int out_size, void* d_ws, size_t ws_size,
                              hipStream_t stream) {
  const float* x  = (const float*)d_in[0];
  const int*  msk = (const int*)  d_in[1];
  const float* wq = (const float*)d_in[2];
  const float* bq = (const float*)d_in[3];
  const float* wk = (const float*)d_in[4];
  const float* bk = (const float*)d_in[5];
  const float* wv = (const float*)d_in[6];
  const float* bv = (const float*)d_in[7];
  const float* wo = (const float*)d_in[8];
  const float* bo = (const float*)d_in[9];
  const float* w1 = (const float*)d_in[10];
  const float* b1 = (const float*)d_in[11];
  const float* w2 = (const float*)d_in[12];
  const float* b2 = (const float*)d_in[13];
  float* out = (float*)d_out;

  size_t off = 0;
  auto alloc = [&](size_t bytes) {
    char* p = (char*)d_ws + off;
    off += (bytes + 255) & ~(size_t)255;
    return (void*)p;
  };
  bf16*  xb    = (bf16*) alloc((size_t)SEQ * DMODEL * 2);
  bf16*  wqkvb = (bf16*) alloc((size_t)DMODEL * DQKV * 2);
  float* bqkv  = (float*)alloc((size_t)DQKV * 4);
  bf16*  wob   = (bf16*) alloc((size_t)DMODEL * DMODEL * 2);
  bf16*  w1b   = (bf16*) alloc((size_t)DMODEL * DFF * 2);
  bf16*  w2b   = (bf16*) alloc((size_t)DFF * DMODEL * 2);
  bf16*  qkvb  = (bf16*) alloc((size_t)SEQ * DQKV * 2);
  bf16*  ctxb  = (bf16*) alloc((size_t)SEQ * DMODEL * 2);
  float* y1f   = (float*)alloc((size_t)SEQ * DMODEL * 4);
  bf16*  y1b   = (bf16*) alloc((size_t)SEQ * DMODEL * 2);
  bf16*  h1b   = (bf16*) alloc((size_t)SEQ * DFF * 2);

  int n;
  n = SEQ * DMODEL;    k_cvt<<<(n + 255) / 256, 256, 0, stream>>>(x, xb, n);
  n = DMODEL * DMODEL; k_pack_qkv<<<(n + 255) / 256, 256, 0, stream>>>(wq, wk, wv, bq, bk, bv, wqkvb, bqkv);
  k_cvt<<<(n + 255) / 256, 256, 0, stream>>>(wo, wob, n);
  n = DMODEL * DFF;    k_cvt<<<(n + 255) / 256, 256, 0, stream>>>(w1, w1b, n);
  k_cvt<<<(n + 255) / 256, 256, 0, stream>>>(w2, w2b, n);

  // QKV = x @ Wqkv + bqkv  (bf16 out)
  k_gemm<false, false, false, true><<<dim3(DQKV / 128, SEQ / 128), 256, 0, stream>>>(
      xb, wqkvb, bqkv, nullptr, nullptr, qkvb, SEQ, DQKV, DMODEL);
  // attention -> ctx (bf16)
  k_attn<<<dim3(SEQ / 128, NH), 256, 0, stream>>>(qkvb, msk, ctxb);
  // y1 = x + ctx @ Wo + bo  (fp32 + bf16 out)
  k_gemm<false, true, true, true><<<dim3(DMODEL / 128, SEQ / 128), 256, 0, stream>>>(
      ctxb, wob, bo, x, y1f, y1b, SEQ, DMODEL, DMODEL);
  // h1 = relu(y1 @ W1 + b1)  (bf16 out)
  k_gemm<true, false, false, true><<<dim3(DFF / 128, SEQ / 128), 256, 0, stream>>>(
      y1b, w1b, b1, nullptr, nullptr, h1b, SEQ, DFF, DMODEL);
  // out = y1 + h1 @ W2 + b2  (fp32 out)
  k_gemm<false, true, true, false><<<dim3(DMODEL / 128, SEQ / 128), 256, 0, stream>>>(
      h1b, w2b, b2, y1f, out, nullptr, SEQ, DMODEL, DFF);
}
